// AdditiveAttention_21792664060216
// MI455X (gfx1250) — matrix operand order repaired and checked by reading, not compiled, not run on hardware
//
#include <hip/hip_runtime.h>

// Problem constants (fixed by the reference)
#define BB   32
#define TT   2048
#define HH   1024            // Q == K == H == 1024
#define ROWS 64              // key rows per workgroup in scores kernel
#define NITER 32             // N-tile iterations per wave (64 tiles / 2 halves)

#define KEYS_BYTES  (ROWS * HH * 2)            // 131072
#define PANEL_BYTES (16 * HH * 2)              // 32768 (one 16-col WkT panel)
#define TQ_OFF      (KEYS_BYTES + 4 * PANEL_BYTES)
#define WV_OFF      (TQ_OFF + HH * 4)
#define SMEM_BYTES  (WV_OFF + HH * 4)          // 270336

typedef __attribute__((ext_vector_type(4)))  __bf16 v4bf;
typedef __attribute__((ext_vector_type(8)))  __bf16 v8bf;
typedef __attribute__((ext_vector_type(16))) __bf16 v16bf;
typedef __attribute__((ext_vector_type(8)))  float  v8f;
typedef unsigned int u32x4 __attribute__((ext_vector_type(4)));
typedef int          i32x8 __attribute__((ext_vector_type(8)));
typedef int          i32x4 __attribute__((ext_vector_type(4)));

__device__ __forceinline__ __bf16 f2bf(float f) {
    unsigned u = __builtin_bit_cast(unsigned, f);
    u += 0x7FFFu + ((u >> 16) & 1u);           // round-to-nearest-even
    unsigned short h = (unsigned short)(u >> 16);
    return __builtin_bit_cast(__bf16, h);
}

__device__ __forceinline__ float fast_tanh(float x) {
    x = fminf(fmaxf(x, -15.0f), 15.0f);
    float e = __expf(2.0f * x);
    return (e - 1.0f) / (e + 1.0f);
}

#if defined(__HIP_DEVICE_COMPILE__) && __has_builtin(__builtin_amdgcn_tensor_load_to_lds) && __has_builtin(__builtin_amdgcn_s_wait_tensorcnt)
#define USE_TDM 1
#else
#define USE_TDM 0
#endif

#if USE_TDM
// One contiguous 32 KB WkT panel -> LDS, via the Tensor Data Mover.
// D# group layouts per CDNA5 ISA ch.8 (count=1, type=2 "image", 1-D tile).
// 6-arg builtin form: (g0, g1, g2, g3, g4, cpol).
__device__ __forceinline__ void tdm_load_panel(const __bf16* gsrc, unsigned lds_byte_off) {
    unsigned long long ga = (unsigned long long)gsrc;
    u32x4 g0;
    g0[0] = 1u;                                   // count=1, is_restore=0, gather=0
    g0[1] = lds_byte_off;                         // lds_addr (bytes)
    g0[2] = (unsigned)ga;                         // global_addr[31:0]
    g0[3] = (unsigned)(ga >> 32) | (2u << 30);    // global_addr[56:32] | type=2
    const int elems = PANEL_BYTES / 2;            // 16384 bf16 elements
    i32x8 g1;
    g1[0] = 1 << 16;                              // wg_mask=0, data_size=1 (2 bytes)
    g1[1] = (elems & 0xFFFF) << 16;               // tensor_dim0[15:0]
    g1[2] = (elems >> 16) & 0xFFFF;               // tensor_dim0[31:16]; tensor_dim1=0
    g1[3] = (elems & 0xFFFF) << 16;               // tile_dim0 = 16384
    g1[4] = 0;                                    // tile_dim1=0 (1-D), tile_dim2=0
    g1[5] = elems;                                // tensor_dim0_stride[31:0]
    g1[6] = 0;
    g1[7] = 0;
    i32x4 z4 = {0, 0, 0, 0};
    i32x8 z8 = {0, 0, 0, 0, 0, 0, 0, 0};
    __builtin_amdgcn_tensor_load_to_lds(g0, g1, z4, z4, z8, 0);
}
#endif

// ---------------------------------------------------------------------------
// Kernel 1: W_k (K x H, row-major fp32) -> WkT (H x K, row-major bf16)
// ---------------------------------------------------------------------------
__global__ void wkT_kernel(const float* __restrict__ Wk, __bf16* __restrict__ WkT) {
    __shared__ float tile[32][33];
    int x  = blockIdx.x * 32 + threadIdx.x;
    int y0 = blockIdx.y * 32;
#pragma unroll
    for (int j = 0; j < 4; ++j)
        tile[threadIdx.y + j * 8][threadIdx.x] = Wk[(size_t)(y0 + threadIdx.y + j * 8) * HH + x];
    __syncthreads();
    int xo = y0 + threadIdx.x;
#pragma unroll
    for (int j = 0; j < 4; ++j) {
        int ho = blockIdx.x * 32 + threadIdx.y + j * 8;
        WkT[(size_t)ho * HH + xo] = f2bf(tile[threadIdx.x][threadIdx.y + j * 8]);
    }
}

// ---------------------------------------------------------------------------
// Kernel 2: trans_q[b,h] = sum_q queries[b,q] * W_q[q,h]
// ---------------------------------------------------------------------------
__global__ void transq_kernel(const float* __restrict__ queries,
                              const float* __restrict__ Wq,
                              float* __restrict__ tq_out) {
    int b = blockIdx.x;
    __shared__ float qv[HH];
    for (int i = threadIdx.x; i < HH; i += 256) qv[i] = queries[b * HH + i];
    __syncthreads();
    float acc[4] = {0.f, 0.f, 0.f, 0.f};
    for (int q = 0; q < HH; ++q) {
        float x = qv[q];
        const float* wr = Wq + (size_t)q * HH;
#pragma unroll
        for (int j = 0; j < 4; ++j) acc[j] += x * wr[threadIdx.x + j * 256];
    }
#pragma unroll
    for (int j = 0; j < 4; ++j) tq_out[b * HH + threadIdx.x + j * 256] = acc[j];
}

// ---------------------------------------------------------------------------
// Kernel 3 (main): fused  scores = tanh(keys@W_k + trans_q) @ w_v
// 128 threads = 4 wave32.  Wave w: M-rows (w&1)*32..+32, N-half (w>>1).
// 2 M-tiles share each B fragment (3 ds_load_b128 per wmma).
// WkT panels DMA'd by the TDM, double buffered, overlapped with compute.
// ---------------------------------------------------------------------------
__global__ void __launch_bounds__(128)
scores_kernel(const float*  __restrict__ keys,
              const __bf16* __restrict__ WkT,
              const float*  __restrict__ transq,
              const float*  __restrict__ w_v,
              float* __restrict__ scores) {
    extern __shared__ __align__(16) char smem[];
    __bf16* sk = (__bf16*)smem;                       // ROWS x HH bf16 (128 KB)
    float*  tq = (float*)(smem + TQ_OFF);
    float*  wv = (float*)(smem + WV_OFF);

    const int b  = blockIdx.x >> 5;                   // 32 slabs per batch
    const int t0 = (blockIdx.x & 31) * ROWS;

    const int lane   = threadIdx.x & 31;
    const int wave   = threadIdx.x >> 5;              // 0..3
    const int mgroup = wave & 1;                      // which 32-row group
    const int nhalf  = wave >> 1;                     // which 32 N-tiles
    const int nl     = lane & 15;
    const int hiHalf = lane >> 4;
    const int koffA  = hiHalf * 8;                    // A: lanes 16-31 hold K 8-15 / 24-31
    const int koffB  = hiHalf * 8;                    // B: lanes 16-31 hold K 8-15 / 24-31

#if USE_TDM
    // Prefetch panels for iterations 0 and 1 (2 panels each: one per N-half).
    if (wave == 0) {
#pragma unroll
        for (int j = 0; j < 2; ++j) {
            tdm_load_panel(WkT + (size_t)(0 + j * 32) * 16 * HH, KEYS_BYTES + (0 * 2 + j) * PANEL_BYTES);
            tdm_load_panel(WkT + (size_t)(1 + j * 32) * 16 * HH, KEYS_BYTES + (1 * 2 + j) * PANEL_BYTES);
        }
    }
#endif

    // ---- stage keys slab -> bf16 LDS (coalesced float4 reads) ----
    {
        const float4* src = (const float4*)(keys + ((size_t)b * TT + t0) * HH);
        for (int i = threadIdx.x; i < ROWS * HH / 4; i += 128) {
            float4 v = src[i];
            v4bf o;
            o[0] = f2bf(v.x); o[1] = f2bf(v.y); o[2] = f2bf(v.z); o[3] = f2bf(v.w);
            *(v4bf*)(sk + (size_t)i * 4) = o;
        }
        for (int i = threadIdx.x; i < HH; i += 128) {
            tq[i] = transq[b * HH + i];
            wv[i] = w_v[i];
        }
    }

    const __bf16* arow0 = sk + (size_t)(mgroup * 32 + nl) * HH;
    const __bf16* arow1 = arow0 + 16 * HH;

    float sacc0[8] = {0.f, 0.f, 0.f, 0.f, 0.f, 0.f, 0.f, 0.f};
    float sacc1[8] = {0.f, 0.f, 0.f, 0.f, 0.f, 0.f, 0.f, 0.f};

    for (int i = 0; i < NITER; ++i) {
        const int buf = i & 1;
#if USE_TDM
        if (wave == 0) {
            if (i < NITER - 1)
                __builtin_amdgcn_s_wait_tensorcnt(2);  // iter i's 2 panels landed
            else
                __builtin_amdgcn_s_wait_tensorcnt(0);  // last iteration: all done
        }
#else
        // Fallback: stage both panels for iteration i into buf with plain loads.
#pragma unroll
        for (int h = 0; h < 2; ++h) {
            const uint4* src = (const uint4*)(WkT + (size_t)(h * 32 + i) * 16 * HH);
            uint4* dst = (uint4*)(smem + KEYS_BYTES + (buf * 2 + h) * PANEL_BYTES);
            for (int j = threadIdx.x; j < PANEL_BYTES / 16; j += 128) dst[j] = src[j];
        }
#endif
        __syncthreads();                              // panels visible to all waves

        const __bf16* brow =
            (const __bf16*)(smem + KEYS_BYTES + (size_t)(buf * 2 + nhalf) * PANEL_BYTES)
            + (size_t)nl * HH;

        v8f acc0 = {}, acc1 = {};
#pragma unroll 4
        for (int k = 0; k < HH; k += 32) {
            v8bf a0lo = *(const v8bf*)(arow0 + k + koffA);
            v8bf a0hi = *(const v8bf*)(arow0 + k + 16 + koffA);
            v16bf a0 = __builtin_shufflevector(a0lo, a0hi,
                         0, 1, 2, 3, 4, 5, 6, 7, 8, 9, 10, 11, 12, 13, 14, 15);
            v8bf a1lo = *(const v8bf*)(arow1 + k + koffA);
            v8bf a1hi = *(const v8bf*)(arow1 + k + 16 + koffA);
            v16bf a1 = __builtin_shufflevector(a1lo, a1hi,
                         0, 1, 2, 3, 4, 5, 6, 7, 8, 9, 10, 11, 12, 13, 14, 15);
            v8bf blo = *(const v8bf*)(brow + k + koffB);
            v8bf bhi = *(const v8bf*)(brow + k + 16 + koffB);
            v16bf bm = __builtin_shufflevector(blo, bhi,
                         0, 1, 2, 3, 4, 5, 6, 7, 8, 9, 10, 11, 12, 13, 14, 15);
            acc0 = __builtin_amdgcn_wmma_f32_16x16x32_bf16(
                       false, a0, false, bm, (short)0, acc0, false, false);
            acc1 = __builtin_amdgcn_wmma_f32_16x16x32_bf16(
                       false, a1, false, bm, (short)0, acc1, false, false);
        }

        const int   h   = (nhalf * NITER + i) * 16 + nl;
        const float tqh = tq[h];
        const float wvh = wv[h];
#pragma unroll
        for (int r = 0; r < 8; ++r) {
            sacc0[r] += fast_tanh(acc0[r] + tqh) * wvh;
            sacc1[r] += fast_tanh(acc1[r] + tqh) * wvh;
        }

        __syncthreads();                              // all waves done reading buf
#if USE_TDM
        if (wave == 0 && i + 2 < NITER) {             // prefetch iteration i+2 into buf
            tdm_load_panel(WkT + (size_t)(0 * 32 + i + 2) * 16 * HH,
                           KEYS_BYTES + (buf * 2 + 0) * PANEL_BYTES);
            tdm_load_panel(WkT + (size_t)(1 * 32 + i + 2) * 16 * HH,
                           KEYS_BYTES + (buf * 2 + 1) * PANEL_BYTES);
        }
#endif
    }

    // reduce over the 16 columns held across each 16-lane group
#pragma unroll
    for (int r = 0; r < 8; ++r) {
        float v0 = sacc0[r], v1 = sacc1[r];
        v0 += __shfl_xor(v0, 1); v1 += __shfl_xor(v1, 1);
        v0 += __shfl_xor(v0, 2); v1 += __shfl_xor(v1, 2);
        v0 += __shfl_xor(v0, 4); v1 += __shfl_xor(v1, 4);
        v0 += __shfl_xor(v0, 8); v1 += __shfl_xor(v1, 8);
        sacc0[r] = v0; sacc1[r] = v1;
    }

    __syncthreads();                                  // panel area free for reuse
    float* part = (float*)(smem + KEYS_BYTES);        // part[2][ROWS]
    if (nl == 0) {
        int base = mgroup * 32 + hiHalf * 8;          // lane0 rows r, lane16 rows 8+r
#pragma unroll
        for (int r = 0; r < 8; ++r) {
            part[nhalf * ROWS + base + r]      = sacc0[r];
            part[nhalf * ROWS + base + 16 + r] = sacc1[r];
        }
    }
    __syncthreads();
    if (threadIdx.x < ROWS)
        scores[(size_t)b * TT + t0 + threadIdx.x] =
            part[threadIdx.x] + part[ROWS + threadIdx.x];
}

// ---------------------------------------------------------------------------
// Kernel 4: masked softmax over T, in place
// ---------------------------------------------------------------------------
__global__ void softmax_kernel(float* __restrict__ attn,
                               const unsigned char* __restrict__ mask) {
    int b = blockIdx.x;
    __shared__ float red[256];
    float local[8];
    float mx = -__builtin_inff();
#pragma unroll
    for (int i = 0; i < 8; ++i) {
        int t = threadIdx.x + i * 256;
        float s = attn[(size_t)b * TT + t];
        if (mask[b * TT + t]) s = -__builtin_inff();
        local[i] = s;
        mx = fmaxf(mx, s);
    }
    red[threadIdx.x] = mx; __syncthreads();
    for (int off = 128; off > 0; off >>= 1) {
        if (threadIdx.x < off) red[threadIdx.x] = fmaxf(red[threadIdx.x], red[threadIdx.x + off]);
        __syncthreads();
    }
    mx = red[0]; __syncthreads();
    float sum = 0.f;
#pragma unroll
    for (int i = 0; i < 8; ++i) { local[i] = __expf(local[i] - mx); sum += local[i]; }
    red[threadIdx.x] = sum; __syncthreads();
    for (int off = 128; off > 0; off >>= 1) {
        if (threadIdx.x < off) red[threadIdx.x] += red[threadIdx.x + off];
        __syncthreads();
    }
    float inv = 1.0f / red[0];
#pragma unroll
    for (int i = 0; i < 8; ++i)
        attn[(size_t)b * TT + threadIdx.x + i * 256] = local[i] * inv;
}

// ---------------------------------------------------------------------------
// Kernel 5: context[b,k] = sum_t keys[b,t,k] * attn[b,t]   (T-split + atomics)
// ---------------------------------------------------------------------------
__global__ void zero_kernel(float* __restrict__ p, int n) {
    int i = blockIdx.x * blockDim.x + threadIdx.x;
    if (i < n) p[i] = 0.f;
}

__global__ void context_kernel(const float* __restrict__ keys,
                               const float* __restrict__ attn,
                               float* __restrict__ ctx) {
    int b = blockIdx.x, kb = blockIdx.y, tb = blockIdx.z;
    __shared__ float sa[512];
    int t0 = tb * 512;
    for (int i = threadIdx.x; i < 512; i += 256) sa[i] = attn[(size_t)b * TT + t0 + i];
    __syncthreads();
    int k = kb * 256 + threadIdx.x;
    const float* kp = keys + ((size_t)b * TT + t0) * HH + k;
    float acc = 0.f;
    for (int t = 0; t < 512; ++t) acc += kp[(size_t)t * HH] * sa[t];
    atomicAdd(&ctx[b * HH + k], acc);
}

// ---------------------------------------------------------------------------
extern "C" void kernel_launch(void* const* d_in, const int* in_sizes, int n_in,
                              void* d_out, int out_size, void* d_ws, size_t ws_size,
                              hipStream_t stream) {
    (void)in_sizes; (void)n_in; (void)out_size; (void)ws_size;
    const float*         queries = (const float*)d_in[0];
    const float*         keys    = (const float*)d_in[1];
    const unsigned char* mask    = (const unsigned char*)d_in[2];
    const float*         W_q     = (const float*)d_in[3];
    const float*         W_k     = (const float*)d_in[4];
    const float*         w_v     = (const float*)d_in[5];

    float* attn = (float*)d_out;                  // B*T attention weights
    float* ctx  = attn + (size_t)BB * TT;         // B*K context

    __bf16* wkT = (__bf16*)d_ws;                  // 1024*1024 bf16 (2 MB)
    float*  tq  = (float*)((char*)d_ws + (size_t)HH * HH * 2);  // 32*1024 fp32

    wkT_kernel<<<dim3(HH / 32, HH / 32), dim3(32, 8), 0, stream>>>(W_k, wkT);
    transq_kernel<<<BB, 256, 0, stream>>>(queries, W_q, tq);

    scores_kernel<<<BB * (TT / ROWS), 128, SMEM_BYTES, stream>>>(keys, wkT, tq, w_v, attn);

    softmax_kernel<<<BB, 256, 0, stream>>>(attn, mask);

    zero_kernel<<<(BB * HH + 255) / 256, 256, 0, stream>>>(ctx, BB * HH);
    context_kernel<<<dim3(BB, HH / 256, TT / 512), 256, 0, stream>>>(keys, attn, ctx);
}
